// MaskedTransformerLayer_32796370272784
// MI455X (gfx1250) — matrix operand order repaired and checked by reading, not compiled, not run on hardware
//
#include <hip/hip_runtime.h>
#include <hip/hip_bf16.h>
#include <math.h>

// Problem constants (from reference): B=4, M=2048, H=1024, I=4096, fp32.
#define BB 4
#define MM_ 2048
#define HH 1024
#define II 4096

typedef __attribute__((ext_vector_type(16))) __bf16 v16bf;
typedef __attribute__((ext_vector_type(8)))  float  v8f;
typedef unsigned short ushort_t;
typedef unsigned long long ull_t;

struct U8x32 { uint4 a, b; };   // 32 bytes == one v16bf fragment

__device__ __forceinline__ unsigned short f2bf(float x) {
    unsigned int u = __builtin_bit_cast(unsigned int, x);
    u += 0x7FFFu + ((u >> 16) & 1u);          // round-to-nearest-even
    return (unsigned short)(u >> 16);
}
__device__ __forceinline__ float bf2f(unsigned short u) {
    return __builtin_bit_cast(float, (unsigned int)u << 16);
}
__device__ __forceinline__ v16bf ldfrag(const unsigned short* p0, const unsigned short* p1) {
    U8x32 u;
    u.a = *(const uint4*)p0;   // ds_load_b128
    u.b = *(const uint4*)p1;   // ds_load_b128
    return __builtin_bit_cast(v16bf, u);
}

// ---------------------------------------------------------------------------
// bf16 WMMA GEMM, async-pipelined.
//   A: bf16 [Mrows][K] (lda), B: bf16 [Ncols][K] (ldb)  -> C[m][n]
// Tile 128x128x32, 256 thr (8 waves, 2x4), wave computes 64x32 (8 WMMAs).
// Staging uses GLOBAL_LOAD_ASYNC_TO_LDS_B128 with double-buffered LDS:
// one s_barrier + one s_wait_asynccnt per K-step.
// mode 0: +bias(p0)                 mode 1: +bias, exact GELU
// mode 2: +bias +residual(p1 fp32)  mode 3: scores (diag p0, *scale, mask pI)
// mode 4: ctx += p0[b,row]*(p1-p2)
// Outputs: Cf (fp32) and/or Cbf (bf16), either may be null.
// ---------------------------------------------------------------------------
__global__ __launch_bounds__(256)
void gemm_bf16_kernel(const ushort_t* __restrict__ A, const ushort_t* __restrict__ Bm,
                      float* __restrict__ Cf, ushort_t* __restrict__ Cbf,
                      int Mrows, int Ncols, int Kdim,
                      int lda, int ldb, int ldc,
                      ull_t sA, ull_t sB, ull_t sC,
                      int mode, float scale,
                      const float* __restrict__ p0, const float* __restrict__ p1,
                      const float* __restrict__ p2, const int* __restrict__ pI)
{
    // [buf][row][k] with row stride 40 (80B, 16B aligned, conflict-free b128)
    __shared__ __align__(16) unsigned short Asm[2][128 * 40];
    __shared__ __align__(16) unsigned short Bsm[2][128 * 40];

    const int bz = blockIdx.z;
    const ushort_t* Ab  = A  + (size_t)bz * sA;
    const ushort_t* Bb_ = Bm + (size_t)bz * sB;

    const int mBase = blockIdx.y * 128;
    const int nBase = blockIdx.x * 128;

    const int tid  = threadIdx.x;
    const int wave = tid >> 5;
    const int lane = tid & 31;
    const int lh   = lane >> 4;
    const int ll   = lane & 15;
    const int wy   = wave >> 2;   // 0..1 : 64-row slab
    const int wx   = wave & 3;    // 0..3 : 32-col slab

    // Per-thread staging coordinates: 512 16B-chunks per 128x32 bf16 tile.
    const int e0 = tid;           // chunk ids e, e+256
    const int r0 = e0 >> 2, c0 = e0 & 3;
    const int e1 = tid + 256;
    const int r1 = e1 >> 2, c1 = e1 & 3;

    const unsigned ldsA[2] = { (unsigned)(uintptr_t)&Asm[0][0], (unsigned)(uintptr_t)&Asm[1][0] };
    const unsigned ldsB[2] = { (unsigned)(uintptr_t)&Bsm[0][0], (unsigned)(uintptr_t)&Bsm[1][0] };

    auto issue = [&](int sel, int k0) {
        unsigned la0 = ldsA[sel] + r0 * 80 + c0 * 16;
        unsigned la1 = ldsA[sel] + r1 * 80 + c1 * 16;
        unsigned lb0 = ldsB[sel] + r0 * 80 + c0 * 16;
        unsigned lb1 = ldsB[sel] + r1 * 80 + c1 * 16;
        ull_t ga0 = (ull_t)(uintptr_t)(Ab  + (size_t)(mBase + r0) * lda + k0 + c0 * 8);
        ull_t ga1 = (ull_t)(uintptr_t)(Ab  + (size_t)(mBase + r1) * lda + k0 + c1 * 8);
        ull_t gb0 = (ull_t)(uintptr_t)(Bb_ + (size_t)(nBase + r0) * ldb + k0 + c0 * 8);
        ull_t gb1 = (ull_t)(uintptr_t)(Bb_ + (size_t)(nBase + r1) * ldb + k0 + c1 * 8);
        asm volatile("global_load_async_to_lds_b128 %0, %1, off" :: "v"(la0), "v"(ga0) : "memory");
        asm volatile("global_load_async_to_lds_b128 %0, %1, off" :: "v"(la1), "v"(ga1) : "memory");
        asm volatile("global_load_async_to_lds_b128 %0, %1, off" :: "v"(lb0), "v"(gb0) : "memory");
        asm volatile("global_load_async_to_lds_b128 %0, %1, off" :: "v"(lb1), "v"(gb1) : "memory");
    };

    v8f acc[4][2] = {};

    const int nIter = Kdim >> 5;
    issue(0, 0);
    for (int it = 0; it < nIter; ++it) {
        asm volatile("s_wait_asynccnt 0" ::: "memory");
        __syncthreads();                       // tile it visible to all waves
        if (it + 1 < nIter) issue((it + 1) & 1, (it + 1) << 5);

        const unsigned short* As = &Asm[it & 1][0];
        const unsigned short* Bs = &Bsm[it & 1][0];

        v16bf af[4], bfv[2];
#pragma unroll
        for (int mt = 0; mt < 4; ++mt) {
            const unsigned short* p = As + (wy * 64 + mt * 16 + ll) * 40 + lh * 8;
            af[mt] = ldfrag(p, p + 16);        // K = lh*8..+7, 16+lh*8..+7
        }
#pragma unroll
        for (int nt = 0; nt < 2; ++nt) {
            const unsigned short* p = Bs + (wx * 32 + nt * 16 + ll) * 40 + lh * 8;
            bfv[nt] = ldfrag(p, p + 16);       // K = lh*8..+7, 16+lh*8..+7
        }
#pragma unroll
        for (int mt = 0; mt < 4; ++mt)
#pragma unroll
            for (int nt = 0; nt < 2; ++nt)
                acc[mt][nt] = __builtin_amdgcn_wmma_f32_16x16x32_bf16(
                    false, af[mt], false, bfv[nt], (short)0, acc[mt][nt], false, false);
        // single barrier per iteration: re-issue into a buffer happens one full
        // iteration after all waves finished their ds_loads from it.
    }

    float*    Cfb = Cf  ? Cf  + (size_t)bz * sC : nullptr;
    ushort_t* Cbb = Cbf ? Cbf + (size_t)bz * sC : nullptr;

#pragma unroll
    for (int mt = 0; mt < 4; ++mt)
#pragma unroll
        for (int nt = 0; nt < 2; ++nt)
#pragma unroll
            for (int r = 0; r < 8; ++r) {
                int grow = mBase + wy * 64 + mt * 16 + r + lh * 8;
                int gcol = nBase + wx * 32 + nt * 16 + ll;
                float v = acc[mt][nt][r];
                if (mode == 0) {
                    v += p0[gcol];
                } else if (mode == 1) {
                    v += p0[gcol];
                    v = 0.5f * v * (1.0f + erff(v * 0.70710678118654752f));
                } else if (mode == 2) {
                    v += p0[gcol] + p1[(size_t)grow * ldc + gcol];
                } else if (mode == 3) {
                    if (gcol == grow) v = p0[(size_t)bz * Mrows + grow];
                    v = v * scale + (1.0f - (float)pI[(size_t)bz * Mrows + grow]) * -1.0e9f;
                } else { // mode 4
                    size_t ix = ((size_t)bz * Mrows + grow) * (size_t)ldc + gcol;
                    v += p0[(size_t)bz * Mrows + grow] * (p1[ix] - p2[ix]);
                }
                size_t ci = (size_t)grow * ldc + gcol;
                if (Cfb) Cfb[ci] = v;
                if (Cbb) Cbb[ci] = f2bf(v);
            }
}

// ---- flat fp32 -> bf16, 8 elements / thread ---------------------------------
__global__ __launch_bounds__(256)
void cvt_bf16_kernel(const float* __restrict__ src, ushort_t* __restrict__ dst, int n8)
{
    int i = blockIdx.x * 256 + threadIdx.x;
    if (i >= n8) return;
    size_t b = (size_t)i * 8;
    float4 a = *(const float4*)(src + b);
    float4 c = *(const float4*)(src + b + 4);
    uint4 o;
    o.x = (unsigned)f2bf(a.x) | ((unsigned)f2bf(a.y) << 16);
    o.y = (unsigned)f2bf(a.z) | ((unsigned)f2bf(a.w) << 16);
    o.z = (unsigned)f2bf(c.x) | ((unsigned)f2bf(c.y) << 16);
    o.w = (unsigned)f2bf(c.z) | ((unsigned)f2bf(c.w) << 16);
    *(uint4*)(dst + b) = o;
}

// ---- fp32 [K][N] -> bf16 [N][K] (transpose + convert), 32x32 tiles ----------
__global__ __launch_bounds__(256)
void transpose_cvt_kernel(const float* __restrict__ src, ushort_t* __restrict__ dst,
                          int K, int N, ull_t sIn, ull_t sOut)
{
    __shared__ float t[32][33];
    const float* s = src + (size_t)blockIdx.z * sIn;
    ushort_t*    d = dst + (size_t)blockIdx.z * sOut;
    int n0 = blockIdx.x * 32, k0 = blockIdx.y * 32;
    int tx = threadIdx.x & 31, ty = threadIdx.x >> 5;   // 32x8
#pragma unroll
    for (int j = 0; j < 32; j += 8)
        t[ty + j][tx] = s[(size_t)(k0 + ty + j) * N + n0 + tx];
    __syncthreads();
#pragma unroll
    for (int j = 0; j < 32; j += 8)
        d[(size_t)(n0 + ty + j) * K + k0 + tx] = f2bf(t[tx][ty + j]);
}

// ---- s_diag[row] = dot(q_bf[row], ku_bf[row]); one wave per row -------------
__global__ __launch_bounds__(256)
void rowdot_bf_kernel(const ushort_t* __restrict__ a, const ushort_t* __restrict__ b,
                      float* __restrict__ out, int Hn)
{
    int wave = threadIdx.x >> 5, lane = threadIdx.x & 31;
    int row = blockIdx.x * 8 + wave;
    const ushort_t* ar = a + (size_t)row * Hn;
    const ushort_t* br = b + (size_t)row * Hn;
    float s = 0.0f;
    for (int c = lane; c < (Hn >> 2); c += 32) {
        uint2 ua = *(const uint2*)(ar + c * 4);
        uint2 ub = *(const uint2*)(br + c * 4);
        s += bf2f(ua.x & 0xffff) * bf2f(ub.x & 0xffff);
        s += bf2f(ua.x >> 16)    * bf2f(ub.x >> 16);
        s += bf2f(ua.y & 0xffff) * bf2f(ub.y & 0xffff);
        s += bf2f(ua.y >> 16)    * bf2f(ub.y >> 16);
    }
#pragma unroll
    for (int o = 16; o > 0; o >>= 1) s += __shfl_down(s, o, 32);
    if (lane == 0) out[row] = s;
}

// ---- row softmax: fp32 scores in -> bf16 probs out + diagonal prob ----------
__global__ __launch_bounds__(256)
void softmax_kernel(const float* __restrict__ s, ushort_t* __restrict__ pr,
                    float* __restrict__ pdiag, int Mn)
{
    __shared__ float red[256];
    int row  = blockIdx.x;          // 0 .. B*M-1
    int dcol = row % Mn;
    const float* sr = s  + (size_t)row * Mn;
    ushort_t*    pw = pr + (size_t)row * Mn;
    int tid = threadIdx.x;

    float mx = -3.0e38f;
    for (int j = tid; j < Mn; j += 256) mx = fmaxf(mx, sr[j]);
    red[tid] = mx; __syncthreads();
    for (int o = 128; o > 0; o >>= 1) { if (tid < o) red[tid] = fmaxf(red[tid], red[tid + o]); __syncthreads(); }
    mx = red[0]; __syncthreads();

    float sum = 0.0f;
    for (int j = tid; j < Mn; j += 256) sum += __expf(sr[j] - mx);
    red[tid] = sum; __syncthreads();
    for (int o = 128; o > 0; o >>= 1) { if (tid < o) red[tid] += red[tid + o]; __syncthreads(); }
    float inv = 1.0f / red[0];

    for (int j = tid; j < Mn; j += 256) {
        float p = __expf(sr[j] - mx) * inv;
        pw[j] = f2bf(p);
        if (j == dcol) pdiag[row] = p;
    }
}

// ---- LayerNorm: fp32 in -> fp32 and/or bf16 out -----------------------------
__global__ __launch_bounds__(256)
void layernorm_kernel(const float* __restrict__ x, const float* __restrict__ g,
                      const float* __restrict__ bta,
                      float* __restrict__ yf, ushort_t* __restrict__ ybf, int Hn)
{
    __shared__ float r1[256], r2[256];
    int row = blockIdx.x;
    const float* xr = x + (size_t)row * Hn;
    int tid = threadIdx.x;
    float s1 = 0.0f, s2 = 0.0f;
    for (int i = tid; i < Hn; i += 256) { float v = xr[i]; s1 += v; s2 += v * v; }
    r1[tid] = s1; r2[tid] = s2; __syncthreads();
    for (int o = 128; o > 0; o >>= 1) {
        if (tid < o) { r1[tid] += r1[tid + o]; r2[tid] += r2[tid + o]; }
        __syncthreads();
    }
    float mean = r1[0] / Hn;
    float var  = r2[0] / Hn - mean * mean;
    float inv  = rsqrtf(var + 1e-12f);
    for (int i = tid; i < Hn; i += 256) {
        float r = (xr[i] - mean) * inv * g[i] + bta[i];
        size_t ix = (size_t)row * Hn + i;
        if (yf)  yf[ix]  = r;
        if (ybf) ybf[ix] = f2bf(r);
    }
}

// ---------------------------------------------------------------------------
extern "C" void kernel_launch(void* const* d_in, const int* in_sizes, int n_in,
                              void* d_out, int out_size, void* d_ws, size_t ws_size,
                              hipStream_t stream)
{
    const float* ht   = (const float*)d_in[0];
    const float* hu   = (const float*)d_in[1];
    const int*   mask = (const int*)  d_in[2];
    const float* Wq = (const float*)d_in[3];  const float* bq = (const float*)d_in[4];
    const float* Wk = (const float*)d_in[5];  const float* bk = (const float*)d_in[6];
    const float* Wv = (const float*)d_in[7];  const float* bv = (const float*)d_in[8];
    const float* Wo = (const float*)d_in[9];  const float* bo = (const float*)d_in[10];
    const float* ln1g = (const float*)d_in[11]; const float* ln1b = (const float*)d_in[12];
    const float* W1 = (const float*)d_in[13]; const float* b1 = (const float*)d_in[14];
    const float* W2 = (const float*)d_in[15]; const float* b2 = (const float*)d_in[16];
    const float* ln2g = (const float*)d_in[17]; const float* ln2b = (const float*)d_in[18];
    float* out = (float*)d_out;

    const size_t MH  = (size_t)BB * MM_ * HH;   // 8.39M
    const size_t MMs = (size_t)BB * MM_ * MM_;  // 16.8M
    const size_t MI  = (size_t)BB * MM_ * II;   // 33.6M
    const int    BM  = BB * MM_;                // 8192

    // Byte-offset workspace allocator (all regions 256B aligned).
    char*  w = (char*)d_ws;
    size_t o = 0;
    auto alloc = [&](size_t bytes) { void* p = w + o; o = (o + bytes + 255) & ~(size_t)255; return p; };

    ushort_t* hu_bf  = (ushort_t*)alloc(MH * 2);
    ushort_t* ht_bf  = (ushort_t*)alloc(MH * 2);
    ushort_t* WqT    = (ushort_t*)alloc((size_t)HH * HH * 2);
    ushort_t* WkT    = (ushort_t*)alloc((size_t)HH * HH * 2);
    ushort_t* WvT    = (ushort_t*)alloc((size_t)HH * HH * 2);
    ushort_t* WoT    = (ushort_t*)alloc((size_t)HH * HH * 2);
    ushort_t* W1T    = (ushort_t*)alloc((size_t)HH * II * 2);
    ushort_t* W2T    = (ushort_t*)alloc((size_t)HH * II * 2);
    ushort_t* q_bf   = (ushort_t*)alloc(MH * 2);
    ushort_t* ku_bf  = (ushort_t*)alloc(MH * 2);
    ushort_t* kt_bf  = (ushort_t*)alloc(MH * 2);
    float*    vu_f   = (float*)   alloc(MH * 4);
    float*    vt_f   = (float*)   alloc(MH * 4);    // reused: attn_f after ctx
    ushort_t* vtT_bf = (ushort_t*)alloc(MH * 2);    // reused: attn_bf after ctx
    float*    sc_f   = (float*)   alloc(MMs * 4);   // reused: ctxWo_f / tmp2_f
    ushort_t* pr_bf  = (ushort_t*)alloc(MMs * 2);
    ushort_t* ctx_bf = (ushort_t*)alloc(MH * 2);
    ushort_t* int_bf = (ushort_t*)alloc(MI * 2);
    float*    sdiag  = (float*)   alloc((size_t)BM * 4);
    float*    pdiag  = (float*)   alloc((size_t)BM * 4);

    float*    ctxWo_f = sc_f;                 // scores dead after softmax
    float*    tmp2_f  = sc_f + MH;            // second 32MB of scores region
    float*    attn_f  = vt_f;                 // vt_f dead after ctx GEMM
    ushort_t* attn_bf = vtT_bf;               // vtT dead after ctx GEMM

    dim3 blk(256);
    const ull_t stMH = (ull_t)MM_ * HH;
    const ull_t stMM = (ull_t)MM_ * MM_;

    dim3 gH  (HH / 128,  BM / 128, 1);
    dim3 gI  (II / 128,  BM / 128, 1);
    dim3 gS  (MM_ / 128, MM_ / 128, BB);
    dim3 gCtx(HH / 128,  MM_ / 128, BB);

    // 0: precision conversion pre-passes
    cvt_bf16_kernel<<<dim3((unsigned)(MH / 8 / 256)), blk, 0, stream>>>(hu, hu_bf, (int)(MH / 8));
    cvt_bf16_kernel<<<dim3((unsigned)(MH / 8 / 256)), blk, 0, stream>>>(ht, ht_bf, (int)(MH / 8));
    dim3 gTW(HH / 32, HH / 32, 1);
    transpose_cvt_kernel<<<gTW, blk, 0, stream>>>(Wq, WqT, HH, HH, 0, 0);
    transpose_cvt_kernel<<<gTW, blk, 0, stream>>>(Wk, WkT, HH, HH, 0, 0);
    transpose_cvt_kernel<<<gTW, blk, 0, stream>>>(Wv, WvT, HH, HH, 0, 0);
    transpose_cvt_kernel<<<gTW, blk, 0, stream>>>(Wo, WoT, HH, HH, 0, 0);
    transpose_cvt_kernel<<<dim3(II / 32, HH / 32, 1), blk, 0, stream>>>(W1, W1T, HH, II, 0, 0); // [I][H]
    transpose_cvt_kernel<<<dim3(HH / 32, II / 32, 1), blk, 0, stream>>>(W2, W2T, II, HH, 0, 0); // [H][I]

    // 1-5: projections
    gemm_bf16_kernel<<<gH, blk, 0, stream>>>(hu_bf, WqT, nullptr, q_bf,  BM, HH, HH, HH, HH, HH, 0,0,0, 0, 1.0f, bq, nullptr, nullptr, nullptr);
    gemm_bf16_kernel<<<gH, blk, 0, stream>>>(hu_bf, WkT, nullptr, ku_bf, BM, HH, HH, HH, HH, HH, 0,0,0, 0, 1.0f, bk, nullptr, nullptr, nullptr);
    gemm_bf16_kernel<<<gH, blk, 0, stream>>>(hu_bf, WvT, vu_f,  nullptr, BM, HH, HH, HH, HH, HH, 0,0,0, 0, 1.0f, bv, nullptr, nullptr, nullptr);
    gemm_bf16_kernel<<<gH, blk, 0, stream>>>(ht_bf, WkT, nullptr, kt_bf, BM, HH, HH, HH, HH, HH, 0,0,0, 0, 1.0f, bk, nullptr, nullptr, nullptr);
    gemm_bf16_kernel<<<gH, blk, 0, stream>>>(ht_bf, WvT, vt_f,  nullptr, BM, HH, HH, HH, HH, HH, 0,0,0, 0, 1.0f, bv, nullptr, nullptr, nullptr);

    // 6: vt^T (per batch) for ctx GEMM B operand; 7: s_diag = q.ku
    transpose_cvt_kernel<<<dim3(HH / 32, MM_ / 32, BB), blk, 0, stream>>>(vt_f, vtT_bf, MM_, HH, stMH, stMH);
    rowdot_bf_kernel<<<dim3(BM / 8), blk, 0, stream>>>(q_bf, ku_bf, sdiag, HH);

    // 8: scores = q @ kt^T (B operand already [n][k]), diag/scale/mask fused
    gemm_bf16_kernel<<<gS, blk, 0, stream>>>(q_bf, kt_bf, sc_f, nullptr, MM_, MM_, HH, HH, HH, MM_,
                                             stMH, stMH, stMM, 3, 0.03125f, sdiag, nullptr, nullptr, mask);

    // 9: softmax -> bf16 probs + p_diag
    softmax_kernel<<<dim3(BM), blk, 0, stream>>>(sc_f, pr_bf, pdiag, MM_);

    // 10: ctx = probs @ vt + p_diag*(vu - vt)
    gemm_bf16_kernel<<<gCtx, blk, 0, stream>>>(pr_bf, vtT_bf, nullptr, ctx_bf, MM_, HH, MM_, MM_, MM_, HH,
                                               stMM, stMH, stMH, 4, 1.0f, pdiag, vu_f, vt_f, nullptr);

    // 11: ctx @ Wo + bo ; 12: LN1 -> attn (fp32 + bf16)
    gemm_bf16_kernel<<<gH, blk, 0, stream>>>(ctx_bf, WoT, ctxWo_f, nullptr, BM, HH, HH, HH, HH, HH, 0,0,0, 0, 1.0f, bo, nullptr, nullptr, nullptr);
    layernorm_kernel<<<dim3(BM), blk, 0, stream>>>(ctxWo_f, ln1g, ln1b, attn_f, attn_bf, HH);

    // 13: inter = gelu(attn @ W1 + b1)
    gemm_bf16_kernel<<<gI, blk, 0, stream>>>(attn_bf, W1T, nullptr, int_bf, BM, II, HH, HH, HH, II, 0,0,0, 1, 1.0f, b1, nullptr, nullptr, nullptr);

    // 14: tmp = inter @ W2 + b2 + attn ; 15: LN2 -> out
    gemm_bf16_kernel<<<gH, blk, 0, stream>>>(int_bf, W2T, tmp2_f, nullptr, BM, HH, II, II, II, HH, 0,0,0, 2, 1.0f, b2, attn_f, nullptr, nullptr);
    layernorm_kernel<<<dim3(BM), blk, 0, stream>>>(tmp2_f, ln2g, ln2b, out, nullptr, HH);
}
